// AttentionConvWrapper_16758962389564
// MI455X (gfx1250) — matrix operand order repaired and checked by reading, not compiled, not run on hardware
//
#include <hip/hip_runtime.h>

typedef __attribute__((ext_vector_type(16))) _Float16 v16h;
typedef __attribute__((ext_vector_type(8)))  float    v8f;
typedef __attribute__((ext_vector_type(4)))  int      v4i;

#define N_NODES   50000
#define N_EDGES   800000
#define F_IN      128
#define HC        64      // H*C
#define NHEAD     4
#define CH        16
#define D_OUT     128
#define KAUG      80      // 64 conv cols + 4 a_src + 4 a_dst + 8 pad
#define NEG_SLOPE 0.2f
#define LN_EPS    1e-5f
#define NT        (N_NODES / 16)   // 3125 wave tiles, exact

#define FRAG1_ELEMS (5 * 4 * 32 * 16)   // gemm1 B fragments: 5 tiles x 4 kchunks
#define FRAG2_ELEMS (8 * 2 * 32 * 16)   // gemm2 B fragments: 8 tiles x 2 kchunks

// ---------- CDNA5 async global->LDS staging (ASYNCcnt) ----------
__device__ __forceinline__ void async_g2l_b128(const void* g, void* l) {
#if __has_builtin(__builtin_amdgcn_global_load_async_to_lds_b128)
  __builtin_amdgcn_global_load_async_to_lds_b128((v4i*)g, (v4i*)l, 0, 0);
#else
  *(float4*)l = *(const float4*)g;
#endif
}
__device__ __forceinline__ void async_wait0() {
#if __has_builtin(__builtin_amdgcn_s_wait_asynccnt)
  __builtin_amdgcn_s_wait_asynccnt(0);
#endif
}

// ---------- order-preserving float<->uint encoding for atomicMax ----------
__device__ __forceinline__ unsigned fenc(float f) {
  unsigned b = __float_as_uint(f);
  return (b & 0x80000000u) ? ~b : (b | 0x80000000u);
}
__device__ __forceinline__ float fdec(unsigned u) {
  unsigned b = (u & 0x80000000u) ? (u & 0x7FFFFFFFu) : ~u;
  return __uint_as_float(b);
}
#define ENC_NEG_INF 0x007FFFFFu  // fenc(-inf)

// ---------- build augmented f16 weight matrix [F_IN x KAUG] ----------
__global__ void k_prep(const float* __restrict__ W,
                       const float* __restrict__ att_src,
                       const float* __restrict__ att_dst,
                       _Float16* __restrict__ Waug) {
  int f = threadIdx.x;
  if (f >= F_IN) return;
  float ws[NHEAD], wd[NHEAD];
#pragma unroll
  for (int h = 0; h < NHEAD; ++h) { ws[h] = 0.f; wd[h] = 0.f; }
#pragma unroll
  for (int h = 0; h < NHEAD; ++h)
#pragma unroll
    for (int c = 0; c < CH; ++c) {
      float w = W[f * HC + h * CH + c];
      Waug[f * KAUG + h * CH + c] = (_Float16)w;
      ws[h] += w * att_src[h * CH + c];
      wd[h] += w * att_dst[h * CH + c];
    }
#pragma unroll
  for (int h = 0; h < NHEAD; ++h) {
    Waug[f * KAUG + 64 + h] = (_Float16)ws[h];
    Waug[f * KAUG + 68 + h] = (_Float16)wd[h];
  }
  for (int c = 72; c < KAUG; ++c) Waug[f * KAUG + c] = (_Float16)0.f;
}

// ---------- pack B operands into per-lane-contiguous WMMA fragment order ----
// frag[t][kc][lane][j] = B[k][col], col = t*16 + lane%16,
//                        k = kc*32 + (lane<16 ? 0 : 8) + (j<8 ? j : j+8)
__global__ void k_pack1(const _Float16* __restrict__ Waug,
                        _Float16* __restrict__ Bfrag1) {
  int i = blockIdx.x * blockDim.x + threadIdx.x;
  if (i >= FRAG1_ELEMS) return;
  int j    = i & 15;
  int lane = (i >> 4) & 31;
  int kc   = (i >> 9) & 3;
  int t    = i >> 11;
  int col  = t * 16 + (lane & 15);
  int k    = kc * 32 + ((lane < 16) ? 0 : 8) + ((j < 8) ? j : j + 8);
  Bfrag1[i] = Waug[k * KAUG + col];
}

__global__ void k_pack2(const float* __restrict__ projw,
                        _Float16* __restrict__ Bfrag2) {
  int i = blockIdx.x * blockDim.x + threadIdx.x;
  if (i >= FRAG2_ELEMS) return;
  int j    = i & 15;
  int lane = (i >> 4) & 31;
  int kc   = (i >> 9) & 1;
  int t    = i >> 10;
  int col  = t * 16 + (lane & 15);
  int k    = kc * 32 + ((lane < 16) ? 0 : 8) + ((j < 8) ? j : j + 8);
  Bfrag2[i] = (_Float16)projw[col * HC + k];   // B[k][col] = proj_w[col,k]
}

// ---------- zero/init accumulation buffers ----------
__global__ void k_init(float* __restrict__ agg, float* __restrict__ denom,
                       unsigned* __restrict__ menc) {
  int i = blockIdx.x * blockDim.x + threadIdx.x;
  if (i < N_NODES * HC) agg[i] = 0.f;
  if (i < N_NODES * NHEAD) { denom[i] = 0.f; menc[i] = ENC_NEG_INF; }
}

// ---------- GEMM1: xp = x @ W  (+ fused a_src/a_dst), B staged in LDS ------
__global__ void k_gemm1(const float* __restrict__ x,
                        const _Float16* __restrict__ Bfrag1,
                        float* __restrict__ xp,
                        float* __restrict__ asrc,
                        float* __restrict__ adst) {
  __shared__ __align__(32) _Float16 ldsB[FRAG1_ELEMS];  // 20 KB
  const int NB128 = FRAG1_ELEMS * 2 / 16;               // 1280 x 16B chunks
  for (int i = threadIdx.x; i < NB128; i += blockDim.x)
    async_g2l_b128(Bfrag1 + i * 8, &ldsB[i * 8]);
  async_wait0();
  __syncthreads();

  int wave = threadIdx.x >> 5;
  int lane = threadIdx.x & 31;
  int tile = blockIdx.x * 4 + wave;
  if (tile >= NT) return;
  int row0  = tile * 16;
  int n     = lane & 15;
  int arow  = row0 + n;                 // A: one row per lane
  int koffA = (lane < 16) ? 0 : 8;      // A lane-half K offset

  v8f acc[5] = {};
  for (int kc = 0; kc < 4; ++kc) {
    const float* xr = x + (size_t)arow * F_IN + kc * 32 + koffA;
    float4 p0 = *(const float4*)(xr);        // K .. K+3
    float4 p1 = *(const float4*)(xr + 4);    // K+4 .. K+7
    float4 p2 = *(const float4*)(xr + 16);   // K+16 .. K+19
    float4 p3 = *(const float4*)(xr + 20);   // K+20 .. K+23
    v16h a;
    a[0]  = (_Float16)p0.x; a[1]  = (_Float16)p0.y;
    a[2]  = (_Float16)p0.z; a[3]  = (_Float16)p0.w;
    a[4]  = (_Float16)p1.x; a[5]  = (_Float16)p1.y;
    a[6]  = (_Float16)p1.z; a[7]  = (_Float16)p1.w;
    a[8]  = (_Float16)p2.x; a[9]  = (_Float16)p2.y;
    a[10] = (_Float16)p2.z; a[11] = (_Float16)p2.w;
    a[12] = (_Float16)p3.x; a[13] = (_Float16)p3.y;
    a[14] = (_Float16)p3.z; a[15] = (_Float16)p3.w;
#pragma unroll
    for (int t = 0; t < 5; ++t) {
      v16h b = *(const v16h*)&ldsB[(((t << 2) + kc) * 32 + lane) * 16];
      acc[t] = __builtin_amdgcn_wmma_f32_16x16x32_f16(
          false, a, false, b, (short)0, acc[t], false, false);
    }
  }

  int rbase = (lane < 16) ? 0 : 8;      // C/D: lanes 16-31 hold rows 8..15
#pragma unroll
  for (int t = 0; t < 4; ++t)
#pragma unroll
    for (int r = 0; r < 8; ++r)
      xp[(size_t)(row0 + rbase + r) * HC + t * 16 + n] = acc[t][r];
#pragma unroll
  for (int r = 0; r < 8; ++r) {
    int row = row0 + rbase + r;
    if (n < 4)      asrc[row * NHEAD + n]       = acc[4][r];
    else if (n < 8) adst[row * NHEAD + (n - 4)] = acc[4][r];
  }
}

// ---------- edge pass 1: leaky-relu logits + segment max ----------
__global__ void k_edge_logits(const int* __restrict__ ei,
                              const float* __restrict__ asrc,
                              const float* __restrict__ adst,
                              float* __restrict__ ebuf,
                              unsigned* __restrict__ menc) {
  int e = blockIdx.x * blockDim.x + threadIdx.x;
  if (e >= N_EDGES) return;
  int s = ei[e], d = ei[N_EDGES + e];
#pragma unroll
  for (int h = 0; h < NHEAD; ++h) {
    float v = asrc[s * NHEAD + h] + adst[d * NHEAD + h];
    v = (v > 0.f) ? v : NEG_SLOPE * v;
    ebuf[(size_t)e * NHEAD + h] = v;
    atomicMax(&menc[d * NHEAD + h], fenc(v));
  }
}

// ---------- edge pass 2: exp(e - m[dst]) + segment sum ----------
__global__ void k_edge_exp(const int* __restrict__ ei,
                           float* __restrict__ ebuf,
                           const unsigned* __restrict__ menc,
                           float* __restrict__ denom) {
  int e = blockIdx.x * blockDim.x + threadIdx.x;
  if (e >= N_EDGES) return;
  int d = ei[N_EDGES + e];
#pragma unroll
  for (int h = 0; h < NHEAD; ++h) {
    float m  = fdec(menc[d * NHEAD + h]);
    float ex = __expf(ebuf[(size_t)e * NHEAD + h] - m);
    ebuf[(size_t)e * NHEAD + h] = ex;
    atomicAdd(&denom[d * NHEAD + h], ex);
  }
}

// ---------- edge pass 3: weighted scatter of messages ----------
__global__ void k_scatter(const int* __restrict__ ei,
                          const float* __restrict__ ebuf,
                          const float* __restrict__ denom,
                          const float* __restrict__ xp,
                          float* __restrict__ agg) {
  int g = blockIdx.x * blockDim.x + threadIdx.x;
  if (g >= N_EDGES * NHEAD) return;
  int e = g >> 2, h = g & 3;
  int s = ei[e], d = ei[N_EDGES + e];
  float alpha = ebuf[(size_t)e * NHEAD + h] / (denom[d * NHEAD + h] + 1e-16f);
  const float* xs = xp  + (size_t)s * HC + h * CH;
  float*       od = agg + (size_t)d * HC + h * CH;
#pragma unroll
  for (int c = 0; c < CH; ++c) atomicAdd(&od[c], alpha * xs[c]);
}

// ---------- GEMM2 (agg+bias)@proj_w^T + proj_b, fused LayerNorm + ReLU ------
__global__ void k_gemm2_ln(const float* __restrict__ agg,
                           const float* __restrict__ bias,
                           const _Float16* __restrict__ Bfrag2,
                           const float* __restrict__ projb,
                           const float* __restrict__ gamma,
                           const float* __restrict__ beta,
                           float* __restrict__ out) {
  __shared__ __align__(32) _Float16 ldsB[FRAG2_ELEMS];  // 16 KB
  const int NB128 = FRAG2_ELEMS * 2 / 16;               // 1024 x 16B chunks
  for (int i = threadIdx.x; i < NB128; i += blockDim.x)
    async_g2l_b128(Bfrag2 + i * 8, &ldsB[i * 8]);
  async_wait0();
  __syncthreads();

  int wave = threadIdx.x >> 5;
  int lane = threadIdx.x & 31;
  int tile = blockIdx.x * 4 + wave;
  if (tile >= NT) return;
  int row0  = tile * 16;
  int n     = lane & 15;
  int arow  = row0 + n;
  int koffA = (lane < 16) ? 0 : 8;

  v8f acc[8] = {};
#pragma unroll
  for (int kc = 0; kc < 2; ++kc) {
    const float* ar = agg  + (size_t)arow * HC + kc * 32 + koffA;
    const float* br = bias + kc * 32 + koffA;
    float4 p0 = *(const float4*)(ar);
    float4 p1 = *(const float4*)(ar + 4);
    float4 p2 = *(const float4*)(ar + 16);
    float4 p3 = *(const float4*)(ar + 20);
    float4 q0 = *(const float4*)(br);
    float4 q1 = *(const float4*)(br + 4);
    float4 q2 = *(const float4*)(br + 16);
    float4 q3 = *(const float4*)(br + 20);
    v16h a;
    a[0]  = (_Float16)(p0.x + q0.x); a[1]  = (_Float16)(p0.y + q0.y);
    a[2]  = (_Float16)(p0.z + q0.z); a[3]  = (_Float16)(p0.w + q0.w);
    a[4]  = (_Float16)(p1.x + q1.x); a[5]  = (_Float16)(p1.y + q1.y);
    a[6]  = (_Float16)(p1.z + q1.z); a[7]  = (_Float16)(p1.w + q1.w);
    a[8]  = (_Float16)(p2.x + q2.x); a[9]  = (_Float16)(p2.y + q2.y);
    a[10] = (_Float16)(p2.z + q2.z); a[11] = (_Float16)(p2.w + q2.w);
    a[12] = (_Float16)(p3.x + q3.x); a[13] = (_Float16)(p3.y + q3.y);
    a[14] = (_Float16)(p3.z + q3.z); a[15] = (_Float16)(p3.w + q3.w);
#pragma unroll
    for (int t = 0; t < 8; ++t) {
      v16h b = *(const v16h*)&ldsB[(((t << 1) + kc) * 32 + lane) * 16];
      acc[t] = __builtin_amdgcn_wmma_f32_16x16x32_f16(
          false, a, false, b, (short)0, acc[t], false, false);
    }
  }

  // + proj_b
#pragma unroll
  for (int t = 0; t < 8; ++t) {
    float pb = projb[t * 16 + n];
#pragma unroll
    for (int r = 0; r < 8; ++r) acc[t][r] += pb;
  }

  // LayerNorm stats: reduce across 16-lane halves (rows r / r+8 independent)
  float s1[8], s2[8];
#pragma unroll
  for (int r = 0; r < 8; ++r) {
    float a1 = 0.f, a2 = 0.f;
#pragma unroll
    for (int t = 0; t < 8; ++t) { float v = acc[t][r]; a1 += v; a2 += v * v; }
    s1[r] = a1; s2[r] = a2;
  }
#pragma unroll
  for (int m = 1; m < 16; m <<= 1) {
#pragma unroll
    for (int r = 0; r < 8; ++r) {
      s1[r] += __shfl_xor(s1[r], m, 32);
      s2[r] += __shfl_xor(s2[r], m, 32);
    }
  }

  int rbase = (lane < 16) ? 0 : 8;
#pragma unroll
  for (int r = 0; r < 8; ++r) {
    float mu  = s1[r] * (1.f / 128.f);
    float var = s2[r] * (1.f / 128.f) - mu * mu;
    float inv = rsqrtf(var + LN_EPS);
    int row = row0 + rbase + r;
#pragma unroll
    for (int t = 0; t < 8; ++t) {
      int col = t * 16 + n;
      float v = (acc[t][r] - mu) * inv * gamma[col] + beta[col];
      out[(size_t)row * D_OUT + col] = fmaxf(v, 0.f);
    }
  }
}

extern "C" void kernel_launch(void* const* d_in, const int* in_sizes, int n_in,
                              void* d_out, int out_size, void* d_ws, size_t ws_size,
                              hipStream_t stream) {
  const float* x       = (const float*)d_in[0];
  const int*   ei      = (const int*)d_in[1];
  const float* W       = (const float*)d_in[2];
  const float* att_src = (const float*)d_in[3];
  const float* att_dst = (const float*)d_in[4];
  const float* bias    = (const float*)d_in[5];
  const float* projw   = (const float*)d_in[6];
  const float* projb   = (const float*)d_in[7];
  const float* gamma   = (const float*)d_in[8];
  const float* beta    = (const float*)d_in[9];
  float* out = (float*)d_out;
  (void)in_sizes; (void)n_in; (void)out_size; (void)ws_size;

  char* ws = (char*)d_ws;
  size_t off = 0;
  auto wsalloc = [&](size_t bytes) -> char* {
    char* p = ws + off;
    off += (bytes + 255) & ~(size_t)255;
    return p;
  };
  _Float16* Waug   = (_Float16*)wsalloc((size_t)F_IN * KAUG * sizeof(_Float16));
  _Float16* Bfrag1 = (_Float16*)wsalloc((size_t)FRAG1_ELEMS * sizeof(_Float16));
  _Float16* Bfrag2 = (_Float16*)wsalloc((size_t)FRAG2_ELEMS * sizeof(_Float16));
  float*    xp     = (float*)wsalloc((size_t)N_NODES * HC * sizeof(float));
  float*    asrc   = (float*)wsalloc((size_t)N_NODES * NHEAD * sizeof(float));
  float*    adst   = (float*)wsalloc((size_t)N_NODES * NHEAD * sizeof(float));
  float*    ebuf   = (float*)wsalloc((size_t)N_EDGES * NHEAD * sizeof(float));
  unsigned* menc   = (unsigned*)wsalloc((size_t)N_NODES * NHEAD * sizeof(unsigned));
  float*    denom  = (float*)wsalloc((size_t)N_NODES * NHEAD * sizeof(float));
  float*    agg    = (float*)wsalloc((size_t)N_NODES * HC * sizeof(float));

  k_prep<<<1, 128, 0, stream>>>(W, att_src, att_dst, Waug);
  k_pack1<<<(FRAG1_ELEMS + 255) / 256, 256, 0, stream>>>(Waug, Bfrag1);
  k_pack2<<<(FRAG2_ELEMS + 255) / 256, 256, 0, stream>>>(projw, Bfrag2);
  k_init<<<(N_NODES * HC + 255) / 256, 256, 0, stream>>>(agg, denom, menc);
  k_gemm1<<<(NT + 3) / 4, 128, 0, stream>>>(x, Bfrag1, xp, asrc, adst);
  k_edge_logits<<<(N_EDGES + 255) / 256, 256, 0, stream>>>(ei, asrc, adst, ebuf, menc);
  k_edge_exp<<<(N_EDGES + 255) / 256, 256, 0, stream>>>(ei, ebuf, menc, denom);
  k_scatter<<<(N_EDGES * NHEAD + 255) / 256, 256, 0, stream>>>(ei, ebuf, denom, xp, agg);
  k_gemm2_ln<<<(NT + 3) / 4, 128, 0, stream>>>(agg, bias, Bfrag2, projb, gamma, beta, out);
}
